// attentionlstm_42064909697681
// MI455X (gfx1250) — matrix operand order repaired and checked by reading, not compiled, not run on hardware
//
#include <hip/hip_runtime.h>
#include <hip/hip_fp16.h>

// ---- problem constants ----
#define Bsz   128
#define Ssz   1024
#define Tsz   512
#define Esz   128
#define HEsz  64
#define HDsz  66
#define ATTsz 32
#define VOC   1000
#define SCHUNK 256
#define NCHUNK 4   // Ssz / SCHUNK

typedef __attribute__((ext_vector_type(16))) _Float16 v16h;
typedef __attribute__((ext_vector_type(8)))  float    v8f;

// Pack two 8-halfs contiguous chunks into a 16-half WMMA operand.
static __device__ __forceinline__ v16h pack2(const _Float16* p0, const _Float16* p1) {
  v16h r;
#pragma unroll
  for (int i = 0; i < 8; i++) { r[i] = p0[i]; r[i + 8] = p1[i]; }
  return r;
}

static __device__ __forceinline__ float red_max16(float v) {
#pragma unroll
  for (int m = 1; m < 16; m <<= 1) v = fmaxf(v, __shfl_xor(v, m, 32));
  return v;
}
static __device__ __forceinline__ float red_sum16(float v) {
#pragma unroll
  for (int m = 1; m < 16; m <<= 1) v += __shfl_xor(v, m, 32);
  return v;
}

// Fast reciprocal: v_rcp_f32 (denominators here are >= 1, well conditioned).
static __device__ __forceinline__ float frcp(float x) {
  return __builtin_amdgcn_rcpf(x);
}

// Hardware tanh on gfx1250 (V_TANH_F32 trans op); branch-free fallback.
static __device__ __forceinline__ float ftanh(float x) {
#if __has_builtin(__builtin_amdgcn_tanhf)
  return __builtin_amdgcn_tanhf(x);
#elif __has_builtin(__builtin_amdgcn_tanh_f32)
  return __builtin_amdgcn_tanh_f32(x);
#else
  // tanh(x) = (e^{2x}-1)/(e^{2x}+1); inputs here are O(1), no overflow risk
  // with v_exp_f32 range, and rcp denominator >= small positive.
  const float t = __expf(2.0f * x);
  return (t - 1.0f) * __builtin_amdgcn_rcpf(t + 1.0f);
#endif
}

static __device__ __forceinline__ float sigm(float x) {
  return frcp(1.0f + __expf(-x));      // v_exp_f32 + v_rcp_f32, branch-free
}

static __device__ __forceinline__ v8f wmma_f16(v16h a, v16h b, v8f c) {
  // D = A(16x32 f16) * B(32x16 f16) + C(16x16 f32)
  return __builtin_amdgcn_wmma_f32_16x16x32_f16(false, a, false, b, (short)0, c, false, false);
}

// ---------------------------------------------------------------------------
// P1: G[v][n] = enc_b[n] + dot(enc_Wih[n,:], emb[v,:])   (1000 x 256 f32)
// Collapses the entire B*S*256*128 encoder input GEMM into a gather table.
// ---------------------------------------------------------------------------
__global__ __launch_bounds__(256) void k_gate_table(
    const float* __restrict__ emb, const float* __restrict__ Wih,
    const float* __restrict__ bvec, float* __restrict__ G) {
  __shared__ float e[Esz];
  const int v = blockIdx.x;
  for (int i = threadIdx.x; i < Esz; i += 256) e[i] = emb[v * Esz + i];
  __syncthreads();
  const int n = threadIdx.x;               // 256 gates
  const float* w = Wih + n * Esz;
  float s = bvec[n];
#pragma unroll 4
  for (int k = 0; k < Esz; k++) s += w[k] * e[k];
  G[v * 256 + n] = s;
}

// P2: enc_Whh -> f16 copy ([256,64], gate-major)
__global__ void k_cvt_whh(const float* __restrict__ Whh, _Float16* __restrict__ Whh_h) {
  int i = blockIdx.x * blockDim.x + threadIdx.x;
  if (i < 256 * HEsz) Whh_h[i] = (_Float16)Whh[i];
}

// P3: a0[b] = att_fc_b + att_fc_W @ emb[0];  hdec = cdec = 0
__global__ void k_init(const float* __restrict__ emb, const float* __restrict__ afcW,
                       const float* __restrict__ afcB, float* __restrict__ a,
                       float* __restrict__ hdec, float* __restrict__ cdec) {
  const int b = blockIdx.x;
  const int j = threadIdx.x;               // 128 threads
  if (j < 64) {
    const float* w = afcW + j * Esz;
    float s = afcB[j];
    for (int k = 0; k < Esz; k++) s += w[k] * emb[k];
    a[b * 64 + j] = s;
  }
  for (int k = j; k < HDsz; k += 128) {
    hdec[b * HDsz + k] = 0.f;
    cdec[b * HDsz + k] = 0.f;
  }
}

// ---------------------------------------------------------------------------
// Encoder LSTM. 8 workgroups x 16 batch rows; 1024 sequential steps.
// Recurrent GEMM h[16,64] @ Whh.T[64,256] via 4 WMMAs/wave/step (8 waves).
// Whh B-fragments preloaded in registers; h kept f16 in LDS; enc out f16.
// ---------------------------------------------------------------------------
__global__ __launch_bounds__(256) void k_encoder(
    const int* __restrict__ x, const float* __restrict__ G,
    const _Float16* __restrict__ Whh_h, _Float16* __restrict__ enc) {
  __shared__ _Float16 h_sm[16][HEsz];
  __shared__ float    c_sm[16][HEsz];
  __shared__ float    g_sm[16][256];
  const int b0 = blockIdx.x * 16;
  const int wave = threadIdx.x >> 5, lane = threadIdx.x & 31;
  const int acol  = lane & 15;
  const int koffB = (lane < 16) ? 0 : 16;  // B layout: 16 contiguous K per lane
  const int koffA = (lane < 16) ? 0 : 16;  // A layout: 16 contiguous K per lane

  // Preload Whh B-fragments: this wave owns gate tiles nt = wave*2 + {0,1}
  v16h bf[2][2];
#pragma unroll
  for (int j = 0; j < 2; j++) {
    const int gate = (wave * 2 + j) * 16 + acol;     // B column (N)
    const _Float16* p = Whh_h + gate * HEsz;
    bf[j][0] = pack2(p + koffB,      p + koffB + 8);       // K 0..31
    bf[j][1] = pack2(p + 32 + koffB, p + 32 + koffB + 8);  // K 32..63
  }
  for (int i = threadIdx.x; i < 16 * HEsz; i += 256) {
    ((_Float16*)h_sm)[i] = (_Float16)0.f;
    ((float*)c_sm)[i] = 0.f;
  }
  __syncthreads();

  for (int s = 0; s < Ssz; s++) {
    // A fragments from h_sm (row = M = acol)
    const _Float16* hrow = &h_sm[acol][0];
    v16h a0 = pack2(hrow + koffA,      hrow + koffA + 8);
    v16h a1 = pack2(hrow + 32 + koffA, hrow + 32 + koffA + 8);
#pragma unroll
    for (int j = 0; j < 2; j++) {
      v8f d = {};
      d = wmma_f16(a0, bf[j][0], d);
      d = wmma_f16(a1, bf[j][1], d);
      const int n = (wave * 2 + j) * 16 + acol;
      const int mbase = (lane < 16) ? 0 : 8;
#pragma unroll
      for (int r = 0; r < 8; r++) g_sm[mbase + r][n] = d[r];
    }
    __syncthreads();
    // Cell update: 16 rows x 64 hidden = 1024 cells, 4 per thread
#pragma unroll
    for (int it = 0; it < 4; ++it) {
      const int cell = threadIdx.x + 256 * it;
      const int m = cell >> 6, hh = cell & 63;
      const int tok = x[(b0 + m) * Ssz + s];
      const float* gr = G + tok * 256;
      const float gi = g_sm[m][hh]        + gr[hh];
      const float gf = g_sm[m][64 + hh]   + gr[64 + hh];
      const float gg = g_sm[m][128 + hh]  + gr[128 + hh];
      const float go = g_sm[m][192 + hh]  + gr[192 + hh];
      const float c  = sigm(gf) * c_sm[m][hh] + sigm(gi) * ftanh(gg);
      const float h  = sigm(go) * ftanh(c);
      c_sm[m][hh] = c;
      const _Float16 hf = (_Float16)h;
      h_sm[m][hh] = hf;
      enc[((size_t)(b0 + m) * Ssz + s) * HEsz + hh] = hf;
    }
    __syncthreads();
  }
}

// ---------------------------------------------------------------------------
// Attention (per decoder step). Grid (chunk=4, b=128). Per workgroup:
//   W'[a,e] = a_b[e] * att_w_W[a,e]  (f16, LDS)  -- folds a⊙enc into weights
//   For each 16-row s-tile: [16,64] x [64,32] via 4 WMMAs, then tanh+softmax
//   over the 32 ATT columns (16-lane shuffle reductions on D fragments) and
//   accumulate partial beta. Deterministic: partials written per (b,chunk).
// ---------------------------------------------------------------------------
__global__ __launch_bounds__(256) void k_attention(
    const _Float16* __restrict__ enc, const float* __restrict__ a,
    const float* __restrict__ attW, const float* __restrict__ attB,
    float* __restrict__ partials) {
  __shared__ _Float16 Wp[ATTsz][HEsz];
  __shared__ float sbeta[8][ATTsz];
  const int chunk = blockIdx.x;            // 0..3  (256 rows of S each)
  const int b = blockIdx.y;                // 0..127
  const float* av = a + b * 64;
  for (int i = threadIdx.x; i < ATTsz * HEsz; i += 256) {
    const int ai = i >> 6, e = i & 63;
    Wp[ai][e] = (_Float16)(av[e] * attW[ai * HEsz + e]);
  }
  __syncthreads();
  const int wave = threadIdx.x >> 5, lane = threadIdx.x & 31;
  const int acol  = lane & 15;
  const int koffB = (lane < 16) ? 0 : 16;
  const int koffA = (lane < 16) ? 0 : 16;
  v16h bfr[2][2];
#pragma unroll
  for (int nt = 0; nt < 2; nt++) {
    const _Float16* p = &Wp[nt * 16 + acol][0];
    bfr[nt][0] = pack2(p + koffB,      p + koffB + 8);
    bfr[nt][1] = pack2(p + 32 + koffB, p + 32 + koffB + 8);
  }
  const float b0v = attB[acol], b1v = attB[16 + acol];
  float acc0 = 0.f, acc1 = 0.f;

  for (int stile = wave; stile < 16; stile += 8) {       // uniform per wave
    const int s0 = chunk * SCHUNK + stile * 16;
    const _Float16* base = enc + ((size_t)b * Ssz + s0 + acol) * HEsz;
    __builtin_prefetch(base + 8 * 16 * HEsz, 0, 1);      // next s-tile
    v16h a0 = pack2(base + koffA,      base + koffA + 8);
    v16h a1 = pack2(base + 32 + koffA, base + 32 + koffA + 8);
    v8f d0 = {}, d1 = {};
    d0 = wmma_f16(a0, bfr[0][0], d0);
    d0 = wmma_f16(a1, bfr[0][1], d0);
    d1 = wmma_f16(a0, bfr[1][0], d1);
    d1 = wmma_f16(a1, bfr[1][1], d1);
#pragma unroll
    for (int r = 0; r < 8; r++) {
      const float v0 = ftanh(d0[r] + b0v);
      const float v1 = ftanh(d1[r] + b1v);
      const float mx = red_max16(fmaxf(v0, v1));
      const float e0 = __expf(v0 - mx), e1 = __expf(v1 - mx);
      const float inv = frcp(red_sum16(e0 + e1));        // sum >= 1 (max term)
      acc0 += e0 * inv;                    // alpha for column a = acol
      acc1 += e1 * inv;                    // alpha for column a = 16+acol
    }
  }
  acc0 += __shfl_xor(acc0, 16, 32);        // combine M halves
  acc1 += __shfl_xor(acc1, 16, 32);
  if (lane < 16) { sbeta[wave][acol] = acc0; sbeta[wave][16 + acol] = acc1; }
  __syncthreads();
  if (threadIdx.x < ATTsz) {
    float s = 0.f;
#pragma unroll
    for (int w = 0; w < 8; w++) s += sbeta[w][threadIdx.x];
    partials[((b * NCHUNK) + chunk) * ATTsz + threadIdx.x] = s;
  }
}

// ---------------------------------------------------------------------------
// Decoder cell (per step). Grid 8 x 16 batch rows. Deterministic beta reduce,
// LSTM cell (HD=66), output write, feedback xemb = h@finW.T, a = xemb@afcW.T.
// ---------------------------------------------------------------------------
__global__ __launch_bounds__(256) void k_decoder(
    const float* __restrict__ partials,
    const float* __restrict__ dWih, const float* __restrict__ dWhh,
    const float* __restrict__ db,   const float* __restrict__ finW,
    const float* __restrict__ finB, const float* __restrict__ afcW,
    const float* __restrict__ afcB, float* __restrict__ hdec,
    float* __restrict__ cdec, float* __restrict__ a,
    float* __restrict__ out, int t) {
  __shared__ float beta[16][ATTsz];
  __shared__ float hsm[16][HDsz];
  __shared__ float gsm[16][4 * HDsz];
  __shared__ float xe[16][Esz];
  const int b0 = blockIdx.x * 16;

  for (int i = threadIdx.x; i < 16 * ATTsz; i += 256) {
    const int m = i >> 5, aa = i & 31;
    float s = 0.f;
#pragma unroll
    for (int c = 0; c < NCHUNK; c++)
      s += partials[(((b0 + m) * NCHUNK) + c) * ATTsz + aa];
    beta[m][aa] = s;
  }
  for (int i = threadIdx.x; i < 16 * HDsz; i += 256) {
    const int m = i / HDsz, k = i % HDsz;
    hsm[m][k] = hdec[(b0 + m) * HDsz + k];
  }
  __syncthreads();
  // gates g[16][264]
  for (int i = threadIdx.x; i < 16 * 4 * HDsz; i += 256) {
    const int m = i / (4 * HDsz), n = i % (4 * HDsz);
    float s = db[n];
    const float* wi = dWih + n * ATTsz;
#pragma unroll
    for (int aa = 0; aa < ATTsz; aa++) s += beta[m][aa] * wi[aa];
    const float* wh = dWhh + n * HDsz;
#pragma unroll 2
    for (int k = 0; k < HDsz; k++) s += hsm[m][k] * wh[k];
    gsm[m][n] = s;
  }
  __syncthreads();
  // cell
  for (int i = threadIdx.x; i < 16 * HDsz; i += 256) {
    const int m = i / HDsz, k = i % HDsz;
    const float gi = gsm[m][k], gf = gsm[m][HDsz + k];
    const float gg = gsm[m][2 * HDsz + k], go = gsm[m][3 * HDsz + k];
    const float c = sigm(gf) * cdec[(b0 + m) * HDsz + k] + sigm(gi) * ftanh(gg);
    const float h = sigm(go) * ftanh(c);
    cdec[(b0 + m) * HDsz + k] = c;
    hdec[(b0 + m) * HDsz + k] = h;
    hsm[m][k] = h;
    out[((size_t)(b0 + m) * Tsz + t) * HDsz + k] = h;
  }
  __syncthreads();
  // xemb = h @ finW.T + finB
  for (int i = threadIdx.x; i < 16 * Esz; i += 256) {
    const int m = i >> 7, e = i & 127;
    float s = finB[e];
    const float* w = finW + e * HDsz;
#pragma unroll 2
    for (int k = 0; k < HDsz; k++) s += hsm[m][k] * w[k];
    xe[m][e] = s;
  }
  __syncthreads();
  // a = xemb @ afcW.T + afcB   (state for next step's attention)
  for (int i = threadIdx.x; i < 16 * 64; i += 256) {
    const int m = i >> 6, j = i & 63;
    float s = afcB[j];
    const float* w = afcW + j * Esz;
#pragma unroll 4
    for (int e = 0; e < Esz; e++) s += xe[m][e] * w[e];
    a[(b0 + m) * 64 + j] = s;
  }
}

// ---------------------------------------------------------------------------
extern "C" void kernel_launch(void* const* d_in, const int* in_sizes, int n_in,
                              void* d_out, int out_size, void* d_ws, size_t ws_size,
                              hipStream_t stream) {
  (void)in_sizes; (void)n_in; (void)out_size; (void)ws_size;
  const int*   x      = (const int*)  d_in[0];
  // d_in[1] = 'c' only provides T; unused numerically
  const float* emb    = (const float*)d_in[2];
  const float* encWih = (const float*)d_in[3];
  const float* encWhh = (const float*)d_in[4];
  const float* encB   = (const float*)d_in[5];
  const float* afcW   = (const float*)d_in[6];
  const float* afcB   = (const float*)d_in[7];
  const float* attW   = (const float*)d_in[8];
  const float* attB   = (const float*)d_in[9];
  const float* dWih   = (const float*)d_in[10];
  const float* dWhh   = (const float*)d_in[11];
  const float* dB     = (const float*)d_in[12];
  const float* finW   = (const float*)d_in[13];
  const float* finB   = (const float*)d_in[14];

  char* ws = (char*)d_ws;
  size_t off = 0;
  float*    G        = (float*)(ws + off);    off += (size_t)VOC * 256 * 4;
  _Float16* Whh_h    = (_Float16*)(ws + off); off += (size_t)256 * HEsz * 2;
  off = (off + 255) & ~(size_t)255;
  _Float16* enc      = (_Float16*)(ws + off); off += (size_t)Bsz * Ssz * HEsz * 2;
  float*    a        = (float*)(ws + off);    off += (size_t)Bsz * 64 * 4;
  float*    hdec     = (float*)(ws + off);    off += (size_t)Bsz * HDsz * 4;
  float*    cdec     = (float*)(ws + off);    off += (size_t)Bsz * HDsz * 4;
  off = (off + 255) & ~(size_t)255;
  float*    partials = (float*)(ws + off);    off += (size_t)Bsz * NCHUNK * ATTsz * 4;
  float* out = (float*)d_out;

  k_gate_table<<<VOC, 256, 0, stream>>>(emb, encWih, encB, G);
  k_cvt_whh<<<64, 256, 0, stream>>>(encWhh, Whh_h);
  k_init<<<Bsz, 128, 0, stream>>>(emb, afcW, afcB, a, hdec, cdec);
  k_encoder<<<Bsz / 16, 256, 0, stream>>>(x, G, Whh_h, enc);
  for (int t = 0; t < Tsz; t++) {
    k_attention<<<dim3(NCHUNK, Bsz), 256, 0, stream>>>(enc, a, attW, attB, partials);
    k_decoder<<<Bsz / 16, 256, 0, stream>>>(partials, dWih, dWhh, dB, finW, finB,
                                            afcW, afcB, hdec, cdec, a, out, t);
  }
}
